// GCNII_Encoder_SubGraph_59425167507616
// MI455X (gfx1250) — matrix operand order repaired and checked by reading, not compiled, not run on hardware
//
#include <hip/hip_runtime.h>
#include <hip/hip_bf16.h>
#include <math.h>

typedef __attribute__((ext_vector_type(16))) __bf16 v16bf;
typedef __attribute__((ext_vector_type(8)))  __bf16 v8bf;
typedef __attribute__((ext_vector_type(8)))  float  v8f;

__device__ __forceinline__ v8f wmma_bf16(v16bf a, v16bf b, v8f c) {
  return __builtin_amdgcn_wmma_f32_16x16x32_bf16(false, a, false, b, (short)0, c,
                                                 false, false);
}

__device__ __forceinline__ v16bf frag_cat(v8bf lo, v8bf hi) {
  return __builtin_shufflevector(lo, hi, 0, 1, 2, 3, 4, 5, 6, 7, 8, 9, 10, 11,
                                 12, 13, 14, 15);
}

// A fragment (16x32 bf16): element e -> K = (e<8?0:16) + (lane>=16?8:0) + (e&7)
__device__ __forceinline__ v16bf load_fragA(const __bf16* rowp, int hiM) {
  v8bf lo = *(const v8bf*)(rowp + hiM);
  v8bf hi = *(const v8bf*)(rowp + hiM + 16);
  return frag_cat(lo, hi);
}

// B fragment (32x16 bf16): element e -> K = (e<8?0:16) + (lane>=16?8:0) + (e&7), N = lane&15
__device__ __forceinline__ v16bf load_fragB(const __bf16* colp) {
  const int h8 = (int)(threadIdx.x & 16) >> 1;
  v8bf lo = *(const v8bf*)(colp - h8);
  v8bf hi = *(const v8bf*)(colp - h8 + 16);
  return frag_cat(lo, hi);
}

// Async global->LDS copy (16B per lane), ASYNCcnt-tracked (ISA 08_async_tensor).
// Generic LDS pointer's low 32 bits == wave-relative LDS address (ISA 10.2).
__device__ __forceinline__ void async_copy_b128(unsigned ldsDst,
                                                const void* gsrc) {
  asm volatile("global_load_async_to_lds_b128 %0, %1, off" ::"v"(ldsDst),
               "v"(gsrc)
               : "memory");
}

__device__ __forceinline__ void wait_async0() {
#if defined(__has_builtin)
#if __has_builtin(__builtin_amdgcn_s_wait_asynccnt)
  __builtin_amdgcn_s_wait_asynccnt(0);
#else
  asm volatile("s_wait_asynccnt 0x0" ::: "memory");
#endif
#else
  asm volatile("s_wait_asynccnt 0x0" ::: "memory");
#endif
}

#define TM 128
#define TN 64
#define TK 32
#define TKP 40  // padded k-stride for transposed B tiles (80B = 5*16B aligned)

// ---------------------------------------------------------------------------
// Generic GEMM: C1 = act(A[M,K] @ B[K,N] + bias), optional copy to C2.
// ---------------------------------------------------------------------------
template <bool RELU, bool STORE2>
__global__ __launch_bounds__(256) void gemm_bias_kernel(
    const float* __restrict__ A, const float* __restrict__ B,
    const float* __restrict__ bias, float* __restrict__ C1,
    float* __restrict__ C2, int M, int N, int K) {
  __shared__ __bf16 sA[TM][TK];
  __shared__ __bf16 sBt[TN][TKP];

  const int tid  = threadIdx.x;
  const int lane = tid & 31, wid = tid >> 5;
  const int waveM = wid >> 1, waveN = wid & 1;  // 4x2 waves -> 128x64 tile
  const int m0 = blockIdx.x * TM;
  const int n0 = blockIdx.y * TN;
  const int hiM = (lane >> 4) * 8;
  const int koff16 = (lane >> 4) * 16;
  const int lrow = lane & 15;

  v8f c[2][2] = {};

  for (int k0 = 0; k0 < K; k0 += TK) {
    if (k0 + TK < K) {  // prefetch next k-tiles into cache
      int pr = tid >> 1, pk = (tid & 1) * 16;
      __builtin_prefetch(&A[(size_t)(m0 + pr) * K + k0 + TK + pk], 0, 3);
      if (tid < 128) {
        int br = tid >> 2, bs = (tid & 3) * 16;
        __builtin_prefetch(&B[(size_t)(k0 + TK + br) * N + n0 + bs], 0, 3);
      }
    }
#pragma unroll 4
    for (int i = tid; i < TM * TK; i += 256) {
      int r = i >> 5, kk = i & 31;
      sA[r][kk] = (__bf16)A[(size_t)(m0 + r) * K + k0 + kk];
    }
#pragma unroll 4
    for (int i = tid; i < TK * TN; i += 256) {
      int kk = i >> 6, nn = i & 63;  // global-coalesced read, transposed store
      sBt[nn][kk] = (__bf16)B[(size_t)(k0 + kk) * N + n0 + nn];
    }
    __syncthreads();

    v16bf af[2], bf[2];
#pragma unroll
    for (int im = 0; im < 2; ++im)
      af[im] = load_fragA(&sA[waveM * 32 + im * 16 + lrow][0], hiM);
#pragma unroll
    for (int in = 0; in < 2; ++in)
      bf[in] = load_fragB(&sBt[waveN * 32 + in * 16 + lrow][koff16]);
#pragma unroll
    for (int im = 0; im < 2; ++im)
#pragma unroll
      for (int in = 0; in < 2; ++in)
        c[im][in] = wmma_bf16(af[im], bf[in], c[im][in]);
    __syncthreads();
  }

#pragma unroll
  for (int im = 0; im < 2; ++im) {
    int rbase = m0 + waveM * 32 + im * 16 + ((lane >> 4) * 8);
#pragma unroll
    for (int in = 0; in < 2; ++in) {
      int col = n0 + waveN * 32 + in * 16 + lrow;
      float bv = bias[col];
#pragma unroll
      for (int j = 0; j < 8; ++j) {
        size_t gi = (size_t)(rbase + j) * N + col;
        float v = c[im][in][j] + bv;
        if (RELU) v = fmaxf(v, 0.0f);
        C1[gi] = v;
        if (STORE2) C2[gi] = v;
      }
    }
  }
}

// ---------------------------------------------------------------------------
// Fused GCNII layer:
// hout = relu( (1-a)agg@(bW1) + a*x0@(bW2) + (1-b)((1-a)agg + a*x0) + hin )
// ---------------------------------------------------------------------------
__global__ __launch_bounds__(256) void gcn2_layer_kernel(
    const float* __restrict__ agg, const float* __restrict__ x0,
    const float* __restrict__ hin, const float* __restrict__ W1,
    const float* __restrict__ W2, float* __restrict__ hout, int M, int H,
    float alpha, float beta) {
  __shared__ __bf16 sA1[TM][TK];
  __shared__ __bf16 sA2[TM][TK];
  __shared__ __bf16 sB1t[TN][TKP];
  __shared__ __bf16 sB2t[TN][TKP];

  const int tid  = threadIdx.x;
  const int lane = tid & 31, wid = tid >> 5;
  const int waveM = wid >> 1, waveN = wid & 1;
  const int m0 = blockIdx.x * TM;
  const int n0 = blockIdx.y * TN;
  const int hiM = (lane >> 4) * 8;
  const int koff16 = (lane >> 4) * 16;
  const int lrow = lane & 15;
  const float sa1 = 1.0f - alpha, sa2 = alpha;

  v8f c[2][2] = {};

  for (int k0 = 0; k0 < H; k0 += TK) {
    if (k0 + TK < H) {
      int pr = tid >> 1, pk = (tid & 1) * 16;
      size_t pa = (size_t)(m0 + pr) * H + k0 + TK + pk;
      __builtin_prefetch(&agg[pa], 0, 3);
      __builtin_prefetch(&x0[pa], 0, 3);
      if (tid < 128) {
        int br = tid >> 2, bs = (tid & 3) * 16;
        size_t pb = (size_t)(k0 + TK + br) * H + n0 + bs;
        __builtin_prefetch(&W1[pb], 0, 3);
        __builtin_prefetch(&W2[pb], 0, 3);
      }
    }
#pragma unroll 4
    for (int i = tid; i < TM * TK; i += 256) {
      int r = i >> 5, kk = i & 31;
      size_t gi = (size_t)(m0 + r) * H + k0 + kk;
      sA1[r][kk] = (__bf16)(sa1 * agg[gi]);
      sA2[r][kk] = (__bf16)(sa2 * x0[gi]);
    }
#pragma unroll 4
    for (int i = tid; i < TK * TN; i += 256) {
      int kk = i >> 6, nn = i & 63;
      size_t gi = (size_t)(k0 + kk) * H + n0 + nn;
      sB1t[nn][kk] = (__bf16)(beta * W1[gi]);
      sB2t[nn][kk] = (__bf16)(beta * W2[gi]);
    }
    __syncthreads();

    v16bf a1f[2], a2f[2], b1f[2], b2f[2];
#pragma unroll
    for (int im = 0; im < 2; ++im) {
      int r = waveM * 32 + im * 16 + lrow;
      a1f[im] = load_fragA(&sA1[r][0], hiM);
      a2f[im] = load_fragA(&sA2[r][0], hiM);
    }
#pragma unroll
    for (int in = 0; in < 2; ++in) {
      int col = waveN * 32 + in * 16 + lrow;
      b1f[in] = load_fragB(&sB1t[col][koff16]);
      b2f[in] = load_fragB(&sB2t[col][koff16]);
    }
#pragma unroll
    for (int im = 0; im < 2; ++im)
#pragma unroll
      for (int in = 0; in < 2; ++in) {
        c[im][in] = wmma_bf16(a1f[im], b1f[in], c[im][in]);
        c[im][in] = wmma_bf16(a2f[im], b2f[in], c[im][in]);
      }
    __syncthreads();
  }

  const float w0 = 1.0f - beta;
#pragma unroll
  for (int im = 0; im < 2; ++im) {
    int rbase = m0 + waveM * 32 + im * 16 + ((lane >> 4) * 8);
#pragma unroll
    for (int in = 0; in < 2; ++in) {
      int col = n0 + waveN * 32 + in * 16 + lrow;
#pragma unroll
      for (int j = 0; j < 8; ++j) {
        size_t gi = (size_t)(rbase + j) * H + col;
        float skip = w0 * (sa1 * agg[gi] + sa2 * x0[gi]);
        float v = c[im][in][j] + skip + hin[gi];
        hout[gi] = fmaxf(v, 0.0f);
      }
    }
  }
}

// ---------------------------------------------------------------------------
// Edge aggregation: agg[dst] += w_e * h[src]   (one wave per edge)
// ---------------------------------------------------------------------------
__global__ __launch_bounds__(256) void aggregate_kernel(
    const float* __restrict__ h, const int* __restrict__ edge_index,
    const float* __restrict__ ew, float* __restrict__ agg, int E, int H) {
  int gw = (int)((blockIdx.x * (size_t)blockDim.x + threadIdx.x) >> 5);
  int lane = threadIdx.x & 31;
  if (gw >= E) return;
  int src = edge_index[gw];
  int dst = edge_index[E + gw];
  float w = ew[gw];
  const float* hs = h + (size_t)src * H;
  float* ad = agg + (size_t)dst * H;
#pragma unroll
  for (int j = lane; j < H; j += 32) atomicAdd(&ad[j], w * hs[j]);
}

__global__ void zero_kernel(float* __restrict__ p, int n) {
  int i = blockIdx.x * blockDim.x + threadIdx.x;
  if (i < n) p[i] = 0.0f;
}

// ---------------------------------------------------------------------------
// Row L2-normalize to bf16: en = emb / max(||emb||, eps)
// ---------------------------------------------------------------------------
__global__ __launch_bounds__(256) void normalize_kernel(
    const float* __restrict__ emb, __bf16* __restrict__ en, int H) {
  __shared__ float red[256];
  int row = blockIdx.x, t = threadIdx.x;
  float v = emb[(size_t)row * H + t];
  red[t] = v * v;
  __syncthreads();
  for (int s = 128; s >= 1; s >>= 1) {
    if (t < s) red[t] += red[t + s];
    __syncthreads();
  }
  float rs = 1.0f / fmaxf(sqrtf(red[0]), 1e-8f);
  en[(size_t)row * H + t] = (__bf16)(v * rs);
}

// ---------------------------------------------------------------------------
// Fused cosine-sim GEMM + masked running top-16 (no materialized NxN matrix).
// Tiles are bf16->bf16 straight copies: use async global->LDS (ASYNCcnt).
// ---------------------------------------------------------------------------
__global__ __launch_bounds__(256) void sim_topk_kernel(
    const __bf16* __restrict__ en, const float* __restrict__ mask,
    float* __restrict__ tkv, int* __restrict__ tki, int Nn, int H) {
  __shared__ __bf16 sA[TM][TK];
  __shared__ __bf16 sBt[TN][TKP];
  __shared__ float sSim[TM][TN + 4];

  const int tid  = threadIdx.x;
  const int lane = tid & 31, wid = tid >> 5;
  const int waveM = wid >> 1, waveN = wid & 1;
  const int m0 = blockIdx.x * TM;
  const int hiM = (lane >> 4) * 8;
  const int koff16 = (lane >> 4) * 16;
  const int lrow = lane & 15;
  const int cr  = tid >> 2;            // 0..63 copy row
  const int seg = (tid & 3) * 8;       // 16B segment within 32-half row

  float tv[16];
  int   ti[16];
#pragma unroll
  for (int k = 0; k < 16; ++k) { tv[k] = -3.0e38f; ti[k] = 0; }

  for (int c0 = 0; c0 < Nn; c0 += TN) {
    v8f c[2][2] = {};
    for (int k0 = 0; k0 < H; k0 += TK) {
      // async tile fills: A = en[m0..m0+127][k0..k0+31], Bt = en[c0..c0+63][..]
      async_copy_b128((unsigned)(size_t)&sA[cr][seg],
                      en + (size_t)(m0 + cr) * H + k0 + seg);
      async_copy_b128((unsigned)(size_t)&sA[cr + 64][seg],
                      en + (size_t)(m0 + cr + 64) * H + k0 + seg);
      async_copy_b128((unsigned)(size_t)&sBt[cr][seg],
                      en + (size_t)(c0 + cr) * H + k0 + seg);
      wait_async0();
      __syncthreads();

      v16bf af[2], bf[2];
#pragma unroll
      for (int im = 0; im < 2; ++im)
        af[im] = load_fragA(&sA[waveM * 32 + im * 16 + lrow][0], hiM);
#pragma unroll
      for (int in = 0; in < 2; ++in)
        bf[in] = load_fragB(&sBt[waveN * 32 + in * 16 + lrow][koff16]);
#pragma unroll
      for (int im = 0; im < 2; ++im)
#pragma unroll
        for (int in = 0; in < 2; ++in)
          c[im][in] = wmma_bf16(af[im], bf[in], c[im][in]);
      __syncthreads();
    }

    // dump masked sim tile to LDS
#pragma unroll
    for (int im = 0; im < 2; ++im) {
      int rl = waveM * 32 + im * 16 + ((lane >> 4) * 8);
#pragma unroll
      for (int in = 0; in < 2; ++in) {
        int cl = waveN * 32 + in * 16 + lrow;
#pragma unroll
        for (int j = 0; j < 8; ++j) {
          float mv = mask[(size_t)(m0 + rl + j) * Nn + (c0 + cl)];
          sSim[rl + j][cl] = c[im][in][j] * mv;
        }
      }
    }
    __syncthreads();

    // per-row running top-16 update
    if (tid < TM) {
      for (int j = 0; j < TN; ++j) {
        float v = sSim[tid][j];
        if (v > tv[15]) {
          int p = 15;
          while (p > 0 && tv[p - 1] < v) {
            tv[p] = tv[p - 1];
            ti[p] = ti[p - 1];
            --p;
          }
          tv[p] = v;
          ti[p] = c0 + j;
        }
      }
    }
    __syncthreads();
  }

  if (tid < TM) {
#pragma unroll
    for (int k = 0; k < 16; ++k) {
      tkv[(size_t)(m0 + tid) * 16 + k] = tv[k];
      tki[(size_t)(m0 + tid) * 16 + k] = ti[k];
    }
  }
}

// ---------------------------------------------------------------------------
// Finalize: out = 0.5*log_softmax(logits) + 0.5*log_softmax(sum_k e^v one_hot)
// ---------------------------------------------------------------------------
__global__ __launch_bounds__(64) void finalize_kernel(
    const float* __restrict__ logits, const float* __restrict__ tkv,
    const int* __restrict__ tki, const int* __restrict__ y,
    float* __restrict__ out, int C) {
  __shared__ float red[64];
  int n = blockIdx.x, c = threadIdx.x;
  float l = logits[(size_t)n * C + c];

  red[c] = l;
  __syncthreads();
  for (int s = 32; s >= 1; s >>= 1) {
    if (c < s) red[c] = fmaxf(red[c], red[c + s]);
    __syncthreads();
  }
  float m1 = red[0];
  __syncthreads();
  red[c] = expf(l - m1);
  __syncthreads();
  for (int s = 32; s >= 1; s >>= 1) {
    if (c < s) red[c] += red[c + s];
    __syncthreads();
  }
  float ls1 = logf(red[0]);
  __syncthreads();
  float plc = l - m1 - ls1;

  float f = 0.0f;
#pragma unroll
  for (int k = 0; k < 16; ++k) {
    int idx = tki[(size_t)n * 16 + k];
    if (y[idx] == c) f += expf(tkv[(size_t)n * 16 + k]);
  }
  red[c] = f;
  __syncthreads();
  for (int s = 32; s >= 1; s >>= 1) {
    if (c < s) red[c] = fmaxf(red[c], red[c + s]);
    __syncthreads();
  }
  float m2 = red[0];
  __syncthreads();
  red[c] = expf(f - m2);
  __syncthreads();
  for (int s = 32; s >= 1; s >>= 1) {
    if (c < s) red[c] += red[c + s];
    __syncthreads();
  }
  float ls2 = logf(red[0]);
  float psim = f - m2 - ls2;

  out[(size_t)n * C + c] = 0.5f * plc + 0.5f * psim;
}

// ---------------------------------------------------------------------------
extern "C" void kernel_launch(void* const* d_in, const int* in_sizes, int n_in,
                              void* d_out, int out_size, void* d_ws,
                              size_t ws_size, hipStream_t stream) {
  const float* x    = (const float*)d_in[0];
  const int*   y    = (const int*)d_in[1];
  const int*   ei   = (const int*)d_in[2];
  const float* ew   = (const float*)d_in[3];
  const float* mask = (const float*)d_in[4];
  const float* W_in = (const float*)d_in[5];
  const float* b_in = (const float*)d_in[6];
  const float* W_o  = (const float*)d_in[7];
  const float* b_o  = (const float*)d_in[8];
  const float* cw1  = (const float*)d_in[9];
  const float* cw2  = (const float*)d_in[10];

  const int Nn  = in_sizes[1];       // 16384
  const int E   = in_sizes[3];       // 524288
  const int Din = in_sizes[0] / Nn;  // 512
  const int H   = in_sizes[6];       // 256
  const int C   = in_sizes[8];       // 64
  const float ALPHA = 0.5f;
  const int NL = 9;

  char* ws = (char*)d_ws;
  float*  hA     = (float*)ws;  ws += (size_t)Nn * H * 4;
  float*  hB     = (float*)ws;  ws += (size_t)Nn * H * 4;
  float*  x0     = (float*)ws;  ws += (size_t)Nn * H * 4;
  float*  agg    = (float*)ws;  ws += (size_t)Nn * H * 4;
  __bf16* en     = (__bf16*)ws; ws += (size_t)Nn * H * 2;
  float*  logits = (float*)ws;  ws += (size_t)Nn * C * 4;
  float*  tkv    = (float*)ws;  ws += (size_t)Nn * 16 * 4;
  int*    tki    = (int*)ws;    ws += (size_t)Nn * 16 * 4;

  // 1) h = relu(x @ W_in + b_in); x0 = h
  gemm_bias_kernel<true, true><<<dim3(Nn / TM, H / TN), 256, 0, stream>>>(
      x, W_in, b_in, hA, x0, Nn, H, Din);

  float* hcur = hA;
  float* hnxt = hB;
  for (int l = 0; l < NL; ++l) {
    zero_kernel<<<(Nn * H + 255) / 256, 256, 0, stream>>>(agg, Nn * H);
    aggregate_kernel<<<(E * 32 + 255) / 256, 256, 0, stream>>>(hcur, ei, ew,
                                                               agg, E, H);
    float beta = logf(1.0f / (float)(l + 1) + 1.0f);  // THETA = 1.0
    gcn2_layer_kernel<<<dim3(Nn / TM, H / TN), 256, 0, stream>>>(
        agg, x0, hcur, cw1 + (size_t)l * H * H, cw2 + (size_t)l * H * H, hnxt,
        Nn, H, ALPHA, beta);
    float* t = hcur; hcur = hnxt; hnxt = t;
  }

  // 2) logits = emb @ W_out + b_out
  gemm_bias_kernel<false, false><<<dim3(Nn / TM, C / TN), 256, 0, stream>>>(
      hcur, W_o, b_o, logits, nullptr, Nn, C, H);

  // 3) row-normalize to bf16
  normalize_kernel<<<Nn, H, 0, stream>>>(hcur, en, H);

  // 4) fused sim GEMM + masked top-16 (async tile fills)
  sim_topk_kernel<<<Nn / TM, 256, 0, stream>>>(en, mask, tkv, tki, Nn, H);

  // 5) fuse + double log_softmax + blend
  finalize_kernel<<<Nn, C, 0, stream>>>(logits, tkv, tki, y, (float*)d_out, C);
}
